// HebbianLayer_20761871909484
// MI455X (gfx1250) — matrix operand order repaired and checked by reading, not compiled, not run on hardware
//
#include <hip/hip_runtime.h>
#include <hip/hip_bf16.h>

typedef __attribute__((ext_vector_type(16))) _Float16 v16h;
typedef __attribute__((ext_vector_type(8)))  _Float16 v8h;
typedef __attribute__((ext_vector_type(4)))  _Float16 v4h;
typedef __attribute__((ext_vector_type(8)))  float    v8f;
typedef __attribute__((ext_vector_type(4)))  float    v4f;
typedef __attribute__((ext_vector_type(4)))  int      v4i;

#define D_HID 512
#define B_ROWS 32768
#define SPLITK 16

// ---------------------------------------------------------------------------
// CDNA5 async global->LDS staging (ASYNCcnt path), with safe fallback
// ---------------------------------------------------------------------------
#if defined(__HIP_DEVICE_COMPILE__) && __has_builtin(__builtin_amdgcn_global_load_async_to_lds_b128)
#define HAVE_ASYNC 1
#else
#define HAVE_ASYNC 0
#endif

#if HAVE_ASYNC
#if __has_builtin(__builtin_amdgcn_s_wait_asynccnt)
#define WAIT_ASYNC() __builtin_amdgcn_s_wait_asynccnt(0)
#else
#define WAIT_ASYNC() asm volatile("s_wait_asynccnt 0x0" ::: "memory")
#endif
#else
#define WAIT_ASYNC()
#endif

// copy 16 bytes global -> LDS (async when available)
__device__ __forceinline__ void cp16(const _Float16* g, _Float16* l) {
#if HAVE_ASYNC
  __builtin_amdgcn_global_load_async_to_lds_b128(
      (__attribute__((address_space(1))) v4i*)g,
      (__attribute__((address_space(3))) v4i*)l, 0, 0);
#else
  *(v8h*)l = *(const v8h*)g;
#endif
}

// ---------------------------------------------------------------------------
// f32 -> f16 convert (vectorized float4 -> half4)
// ---------------------------------------------------------------------------
__global__ __launch_bounds__(256) void cvt_f32_f16(const float* __restrict__ in,
                                                   _Float16* __restrict__ out, int n4) {
  int i = blockIdx.x * blockDim.x + threadIdx.x;
  if (i >= n4) return;
  v4f x = ((const v4f*)in)[i];
  v4h y;
  y[0] = (_Float16)x[0]; y[1] = (_Float16)x[1];
  y[2] = (_Float16)x[2]; y[3] = (_Float16)x[3];
  ((v4h*)out)[i] = y;
}

// transpose + convert: out[n*512 + k] = (f16) in[k*512 + n]   (512x512)
__global__ __launch_bounds__(256) void cvt_t_f32_f16(const float* __restrict__ in,
                                                     _Float16* __restrict__ out) {
  int idx = blockIdx.x * blockDim.x + threadIdx.x;   // 262144 threads
  int n = idx >> 9, k = idx & 511;
  out[idx] = (_Float16)in[k * D_HID + n];
}

// ---------------------------------------------------------------------------
// WMMA fragment helpers (layouts per CDNA5 ISA 7.12.2, wave32)
// ---------------------------------------------------------------------------
__device__ __forceinline__ v16h comb16(v8h lo, v8h hi) {
  v16h r;
#pragma unroll
  for (int i = 0; i < 8; i++) { r[i] = lo[i]; r[i + 8] = hi[i]; }
  return r;
}

// As: [rows][32] halves, row-major (M x K).  A-frag for rows [mbase, mbase+16).
__device__ __forceinline__ v16h load_afrag(const _Float16* As, int mbase, int lane) {
  int m  = mbase + (lane & 15);
  int kh = (lane < 16) ? 0 : 8;
  v8h lo = *(const v8h*)&As[m * 32 + kh];
  v8h hi = *(const v8h*)&As[m * 32 + kh + 16];
  return comb16(lo, hi);
}

// Bs: [cols][32] halves = B^T tile (N x K).  B-frag for cols [nbase, nbase+16).
__device__ __forceinline__ v16h load_bfrag(const _Float16* Bs, int nbase, int lane) {
  int n  = nbase + (lane & 15);
  int kb = (lane < 16) ? 0 : 8;
  v8h lo = *(const v8h*)&Bs[n * 32 + kb];
  v8h hi = *(const v8h*)&Bs[n * 32 + kb + 16];
  return comb16(lo, hi);
}

// ---------------------------------------------------------------------------
// NN GEMM:  C[b][n] = sum_k A[b][k] * Bt[n][k]   (A: 32768x512 f16, Bt: 512x512)
// block tile 256x128x32, 8 waves (4x2), wave tile 64x64 (16 WMMA / k-step),
// LDS double-buffered, async staging.
// ---------------------------------------------------------------------------
__global__ __launch_bounds__(256) void gemm_nn_f16(const _Float16* __restrict__ A,
                                                   const _Float16* __restrict__ Bt,
                                                   _Float16* __restrict__ Ch) {
  __shared__ __align__(16) _Float16 As[2][256 * 32];   // 32 KB
  __shared__ __align__(16) _Float16 Bs[2][128 * 32];   // 16 KB
  const int tid  = threadIdx.x;
  const int lane = tid & 31;
  const int wid  = tid >> 5;
  const int wm   = wid & 3;    // 4 waves over 256 rows (64 each)
  const int wn   = wid >> 2;   // 2 waves over 128 cols (64 each)
  const int row0 = (blockIdx.x >> 2) * 256;
  const int col0 = (blockIdx.x & 3) * 128;

  v8f acc[4][4];
#pragma unroll
  for (int i = 0; i < 4; i++)
#pragma unroll
    for (int j = 0; j < 4; j++) { v8f z = {}; acc[i][j] = z; }

  auto stage = [&](int buf, int k0) {
#pragma unroll
    for (int c = tid; c < 1024; c += 256) {          // A: 256 rows x 4 chunks of 8
      int r = c >> 2, off = (c & 3) * 8;
      cp16(&A[(size_t)(row0 + r) * D_HID + k0 + off], &As[buf][r * 32 + off]);
    }
#pragma unroll
    for (int c = tid; c < 512; c += 256) {           // B: 128 cols x 4 chunks of 8
      int r = c >> 2, off = (c & 3) * 8;
      cp16(&Bt[(size_t)(col0 + r) * D_HID + k0 + off], &Bs[buf][r * 32 + off]);
    }
  };

  const int NK = D_HID / 32;                         // 16 k-steps
  stage(0, 0);
  WAIT_ASYNC();
  __syncthreads();

  for (int kt = 0; kt < NK; kt++) {
    const int cur = kt & 1;
    if (kt + 1 < NK) stage(cur ^ 1, (kt + 1) * 32);

    v16h af[4], bf[4];
#pragma unroll
    for (int i = 0; i < 4; i++) af[i] = load_afrag(&As[cur][0], wm * 64 + i * 16, lane);
#pragma unroll
    for (int j = 0; j < 4; j++) bf[j] = load_bfrag(&Bs[cur][0], wn * 64 + j * 16, lane);
#pragma unroll
    for (int i = 0; i < 4; i++)
#pragma unroll
      for (int j = 0; j < 4; j++)
        acc[i][j] = __builtin_amdgcn_wmma_f32_16x16x32_f16(
            false, af[i], false, bf[j], (short)0, acc[i][j], false, false);

    WAIT_ASYNC();
    __syncthreads();
  }

  // epilogue: C/D layout (VGPR r -> M=r(+8 for upper lanes), N=lane%16) -> f16
#pragma unroll
  for (int i = 0; i < 4; i++)
#pragma unroll
    for (int j = 0; j < 4; j++) {
      int mbase = row0 + wm * 64 + i * 16 + ((lane < 16) ? 0 : 8);
      int n     = col0 + wn * 64 + j * 16 + (lane & 15);
#pragma unroll
      for (int r = 0; r < 8; r++)
        Ch[(size_t)(mbase + r) * D_HID + n] = (_Float16)acc[i][j][r];
    }
}

// ---------------------------------------------------------------------------
// TN GEMM (split-K, deterministic partials):
//   Ppart[sk][i][j] = sum_{b in chunk sk} A[b][i] * B[b][j]
// block tile 128x128x32, transpose during LDS staging, double-buffered.
// grid = 16 tiles * SPLITK blocks of 256 threads.
// ---------------------------------------------------------------------------
__global__ __launch_bounds__(256) void gemm_tn_f16(const _Float16* __restrict__ A,
                                                   const _Float16* __restrict__ B,
                                                   float* __restrict__ Ppart) {
  __shared__ __align__(16) _Float16 As[2][128 * 32];   // [m=i][k=b]
  __shared__ __align__(16) _Float16 Bs[2][128 * 32];   // [n=j][k=b]
  const int tid  = threadIdx.x;
  const int lane = tid & 31;
  const int wid  = tid >> 5;
  const int wm   = wid & 3;
  const int wn   = wid >> 2;
  const int tile = blockIdx.x & 15;
  const int sk   = blockIdx.x >> 4;
  const int i0   = (tile >> 2) * 128, j0 = (tile & 3) * 128;
  const int kchunk = B_ROWS / SPLITK;                  // 2048
  const int kbeg = sk * kchunk;
  const int NK   = kchunk / 32;                        // 64 k-steps

  v8f acc[2][4];
#pragma unroll
  for (int i = 0; i < 2; i++)
#pragma unroll
    for (int j = 0; j < 4; j++) { v8f z = {}; acc[i][j] = z; }

  auto stage = [&](int buf, int k0) {
#pragma unroll
    for (int c = tid; c < 512; c += 256) {             // 32 b-rows x 16 chunks of 8
      int k  = c >> 4;
      int m0 = (c & 15) * 8;
      v8h va = *(const v8h*)&A[(size_t)(k0 + k) * D_HID + i0 + m0];
      v8h vb = *(const v8h*)&B[(size_t)(k0 + k) * D_HID + j0 + m0];
#pragma unroll
      for (int t = 0; t < 8; t++) {                    // transpose into LDS
        As[buf][(m0 + t) * 32 + k] = va[t];
        Bs[buf][(m0 + t) * 32 + k] = vb[t];
      }
    }
  };

  stage(0, kbeg);
  __syncthreads();

  for (int kt = 0; kt < NK; kt++) {
    const int cur = kt & 1;
    if (kt + 1 < NK) stage(cur ^ 1, kbeg + (kt + 1) * 32);

    v16h af[2], bf[4];
#pragma unroll
    for (int i = 0; i < 2; i++) af[i] = load_afrag(&As[cur][0], wm * 32 + i * 16, lane);
#pragma unroll
    for (int j = 0; j < 4; j++) bf[j] = load_bfrag(&Bs[cur][0], wn * 64 + j * 16, lane);
#pragma unroll
    for (int i = 0; i < 2; i++)
#pragma unroll
      for (int j = 0; j < 4; j++)
        acc[i][j] = __builtin_amdgcn_wmma_f32_16x16x32_f16(
            false, af[i], false, bf[j], (short)0, acc[i][j], false, false);

    __syncthreads();
  }

  float* Pk = Ppart + (size_t)sk * (D_HID * D_HID);
#pragma unroll
  for (int i = 0; i < 2; i++)
#pragma unroll
    for (int j = 0; j < 4; j++) {
      int mbase = i0 + wm * 32 + i * 16 + ((lane < 16) ? 0 : 8);
      int n     = j0 + wn * 64 + j * 16 + (lane & 15);
#pragma unroll
      for (int r = 0; r < 8; r++)
        Pk[(size_t)(mbase + r) * D_HID + n] = acc[i][j][r];
    }
}

// ---------------------------------------------------------------------------
// fused:  final = LN_act( relu( stim_out + LN_rec(rec_out) ) )   one block/row
// ---------------------------------------------------------------------------
__device__ __forceinline__ void red2(float& s, float& q, float* sh1, float* sh2) {
  int tid = threadIdx.x;
  sh1[tid] = s; sh2[tid] = q;
  __syncthreads();
  for (int off = 128; off > 0; off >>= 1) {
    if (tid < off) { sh1[tid] += sh1[tid + off]; sh2[tid] += sh2[tid + off]; }
    __syncthreads();
  }
  float rs = sh1[0], rq = sh2[0];
  __syncthreads();
  s = rs; q = rq;
}

__global__ __launch_bounds__(256) void ln_fuse(const _Float16* __restrict__ rec,
                                               const _Float16* __restrict__ stim,
                                               const float* __restrict__ grec,
                                               const float* __restrict__ brec,
                                               const float* __restrict__ gact,
                                               const float* __restrict__ bact,
                                               float* __restrict__ out) {
  __shared__ float sh1[256];
  __shared__ float sh2[256];
  const int tid = threadIdx.x;
  const size_t base = (size_t)blockIdx.x * D_HID;
  float x0 = (float)rec[base + tid], x1 = (float)rec[base + tid + 256];
  float s = x0 + x1, q = x0 * x0 + x1 * x1;
  red2(s, q, sh1, sh2);
  float mu = s * (1.0f / D_HID);
  float rs = rsqrtf(q * (1.0f / D_HID) - mu * mu + 1e-5f);
  float r0 = (x0 - mu) * rs * grec[tid] + brec[tid];
  float r1 = (x1 - mu) * rs * grec[tid + 256] + brec[tid + 256];
  float p0 = fmaxf((float)stim[base + tid] + r0, 0.0f);
  float p1 = fmaxf((float)stim[base + tid + 256] + r1, 0.0f);
  s = p0 + p1; q = p0 * p0 + p1 * p1;
  red2(s, q, sh1, sh2);
  mu = s * (1.0f / D_HID);
  rs = rsqrtf(q * (1.0f / D_HID) - mu * mu + 1e-5f);
  out[base + tid]       = (p0 - mu) * rs * gact[tid] + bact[tid];
  out[base + tid + 256] = (p1 - mu) * rs * gact[tid + 256] + bact[tid + 256];
}

// ---------------------------------------------------------------------------
// weight update:  row := l2norm( W*(1-decay[row]) + (sum_sk Ppart)*alpha[col] )
// one block per row (512 blocks).
// ---------------------------------------------------------------------------
__global__ __launch_bounds__(256) void wupd(const float* __restrict__ Wsrc,
                                            const float* __restrict__ Ppart,
                                            const float* __restrict__ alpha,
                                            const float* __restrict__ decay,
                                            float* __restrict__ out) {
  __shared__ float sh[256];
  const int row = blockIdx.x, tid = threadIdx.x;
  const float dr = 1.0f - decay[row];
  float u0 = 0.0f, u1 = 0.0f;
#pragma unroll
  for (int s = 0; s < SPLITK; s++) {
    const float* Pk = Ppart + (size_t)s * (D_HID * D_HID) + (size_t)row * D_HID;
    u0 += Pk[tid];
    u1 += Pk[tid + 256];
  }
  float t0 = Wsrc[row * D_HID + tid]       * dr + u0 * alpha[tid];
  float t1 = Wsrc[row * D_HID + tid + 256] * dr + u1 * alpha[tid + 256];
  sh[tid] = t0 * t0 + t1 * t1;
  __syncthreads();
  for (int off = 128; off > 0; off >>= 1) {
    if (tid < off) sh[tid] += sh[tid + off];
    __syncthreads();
  }
  float inv = 1.0f / fmaxf(sqrtf(sh[0]), 1e-12f);
  out[row * D_HID + tid]       = t0 * inv;
  out[row * D_HID + tid + 256] = t1 * inv;
}

// ---------------------------------------------------------------------------
extern "C" void kernel_launch(void* const* d_in, const int* in_sizes, int n_in,
                              void* d_out, int out_size, void* d_ws, size_t ws_size,
                              hipStream_t stream) {
  const float* S     = (const float*)d_in[0];
  const float* P     = (const float*)d_in[1];
  const float* W     = (const float*)d_in[2];
  const float* Wr    = (const float*)d_in[3];
  const float* alpha = (const float*)d_in[4];
  const float* decay = (const float*)d_in[5];
  const float* gact  = (const float*)d_in[6];
  const float* bact  = (const float*)d_in[7];
  const float* grec  = (const float*)d_in[8];
  const float* brec  = (const float*)d_in[9];
  float* out = (float*)d_out;

  char* ws = (char*)d_ws;
  const size_t BD2 = (size_t)B_ROWS * D_HID * 2;      // f16 bytes of a [B,D] array
  _Float16* Sh  = (_Float16*)(ws);                    // 32 MB
  _Float16* Ph  = (_Float16*)(ws + BD2);              // 32 MB
  _Float16* Wt  = (_Float16*)(ws + 2 * BD2);          // 0.5 MB
  _Float16* Wrt = (_Float16*)(ws + 2 * BD2 + 524288); // 0.5 MB
  _Float16* Oh  = (_Float16*)(ws + 2 * BD2 + 1048576);           // stim_out f16, 32 MB
  _Float16* Rh  = (_Float16*)(ws + 3 * BD2 + 1048576);           // rec_out  f16, 32 MB
  float*    P1  = (float*)(ws + 4 * BD2 + 1048576);              // split-K partials, 16 MB
  float*    P2  = (float*)(ws + 4 * BD2 + 1048576 + (size_t)SPLITK * D_HID * D_HID * 4);

  const int n4 = B_ROWS * D_HID / 4;
  cvt_f32_f16<<<n4 / 256, 256, 0, stream>>>(S, Sh, n4);
  cvt_f32_f16<<<n4 / 256, 256, 0, stream>>>(P, Ph, n4);
  cvt_t_f32_f16<<<(D_HID * D_HID) / 256, 256, 0, stream>>>(W, Wt);
  cvt_t_f32_f16<<<(D_HID * D_HID) / 256, 256, 0, stream>>>(Wr, Wrt);

  // forward GEMMs: 128 M-tiles x 4 N-tiles (256x128 block tile)
  gemm_nn_f16<<<(B_ROWS / 256) * (D_HID / 128), 256, 0, stream>>>(Sh, Wt, Oh);
  gemm_nn_f16<<<(B_ROWS / 256) * (D_HID / 128), 256, 0, stream>>>(Ph, Wrt, Rh);

  // Hebbian outer products: 16 tiles x SPLITK
  gemm_tn_f16<<<16 * SPLITK, 256, 0, stream>>>(Sh, Oh, P1);
  gemm_tn_f16<<<16 * SPLITK, 256, 0, stream>>>(Ph, Rh, P2);

  // fused LN/relu/LN -> final
  ln_fuse<<<B_ROWS, 256, 0, stream>>>(Rh, Oh, grec, brec, gact, bact, out);

  // weight updates -> new_W, new_Wr
  wupd<<<D_HID, 256, 0, stream>>>(W,  P1, alpha, decay, out + (size_t)B_ROWS * D_HID);
  wupd<<<D_HID, 256, 0, stream>>>(Wr, P2, alpha, decay, out + (size_t)B_ROWS * D_HID + D_HID * D_HID);
}
